// DA3CrossFrameCFAngleLoss_3350074491450
// MI455X (gfx1250) — matrix operand order repaired and checked by reading, not compiled, not run on hardware
//
#include <hip/hip_runtime.h>
#include <hip/hip_bf16.h>

// ---------------- problem constants ----------------
constexpr int BATCH = 2;
constexpr int V     = 8;     // teacher frames
constexpr int P     = 2048;  // points
constexpr int D     = 1024;  // feature dim
constexpr int R     = 512;   // ref_perm size
constexpr int S     = 512;   // shared_perm size
constexpr int TK    = 4;     // TOPK
constexpr int E     = 4 * P; // extra rows = 8192
constexpr int NF    = 3;     // student frames 1..3 (teacher frames 2,4,6)
constexpr int ANGLE_BLOCKS = (BATCH * R * S) / 256; // 2048
constexpr int NPART = NF * ANGLE_BLOCKS;            // 6144
constexpr float EPSF = 1e-8f;

typedef __attribute__((ext_vector_type(16))) __bf16 v16bf;
typedef __attribute__((ext_vector_type(8)))  __bf16 v8bf;
typedef __attribute__((ext_vector_type(8)))  float  v8f;

// ---------------- helpers ----------------
__device__ __forceinline__ float block_reduce_sum(float v, float* sm) {
  int t = threadIdx.x;
  sm[t] = v;
  __syncthreads();
  for (int w = 128; w > 0; w >>= 1) {
    if (t < w) sm[t] += sm[t + w];
    __syncthreads();
  }
  float r = sm[0];
  __syncthreads();
  return r;
}

// ---------------- gather / convert kernels ----------------
__global__ void prep_ref_kernel(const float* __restrict__ teacher,
                                const float* __restrict__ student,
                                const int* __restrict__ ref_perm,
                                __bf16* __restrict__ ref_t_bf,
                                __bf16* __restrict__ ref_s_bf,
                                __bf16* __restrict__ ref_n_bf,
                                float* __restrict__ rr_t,
                                float* __restrict__ rr_s) {
  __shared__ float sm[256];
  int br = blockIdx.x;
  int b = br / R, r = br - b * R;
  int p = ref_perm[r];
  const float* trow = teacher + (((long)b * V + 0) * P + p) * D;
  const float* srow = student + (((long)b * 4 + 0) * P + p) * D;
  long obase = ((long)b * R + r) * D;
  float at = 0.f, as = 0.f;
  for (int d = threadIdx.x; d < D; d += 256) {
    float tv = trow[d], sv = srow[d];
    at += tv * tv;
    as += sv * sv;
    ref_t_bf[obase + d] = (__bf16)tv;
    ref_s_bf[obase + d] = (__bf16)sv;
  }
  float tsum = block_reduce_sum(at, sm);
  float ssum = block_reduce_sum(as, sm);
  if (threadIdx.x == 0) { rr_t[b * R + r] = tsum; rr_s[b * R + r] = ssum; }
  float inv = 1.0f / fmaxf(sqrtf(tsum), 1e-12f);
  for (int d = threadIdx.x; d < D; d += 256)
    ref_n_bf[obase + d] = (__bf16)(trow[d] * inv);
}

__global__ void prep_shared_kernel(const float* __restrict__ teacher,
                                   const float* __restrict__ student,
                                   const int* __restrict__ shared_perm,
                                   __bf16* __restrict__ sh_t_bf,
                                   __bf16* __restrict__ sh_s_bf,
                                   float* __restrict__ ss_t,
                                   float* __restrict__ ss_s) {
  __shared__ float sm[256];
  int id = blockIdx.x;                 // f*BATCH*S + b*S + s
  int f = id / (BATCH * S);
  int rem = id - f * BATCH * S;
  int b = rem / S, s = rem - b * S;
  int p = shared_perm[s];
  int tfr = 2 * (f + 1);               // teacher frames 2,4,6
  const float* trow = teacher + (((long)b * V + tfr) * P + p) * D;
  const float* srow = student + (((long)b * 4 + (f + 1)) * P + p) * D;
  long obase = (((long)f * BATCH + b) * S + s) * D;
  float at = 0.f, as = 0.f;
  for (int d = threadIdx.x; d < D; d += 256) {
    float tv = trow[d], sv = srow[d];
    at += tv * tv;
    as += sv * sv;
    sh_t_bf[obase + d] = (__bf16)tv;
    sh_s_bf[obase + d] = (__bf16)sv;
  }
  float ts = block_reduce_sum(at, sm);
  float ssum = block_reduce_sum(as, sm);
  if (threadIdx.x == 0) {
    ss_t[(f * BATCH + b) * S + s] = ts;
    ss_s[(f * BATCH + b) * S + s] = ssum;
  }
}

__global__ void prep_ext_kernel(const float* __restrict__ teacher,
                                __bf16* __restrict__ ext_n_bf) {
  __shared__ float sm[256];
  int id = blockIdx.x;                 // b*E + e
  int b = id / E, e = id - b * E;
  int fr = 1 + 2 * (e >> 11);          // EXTRA_FRAMES = 1,3,5,7
  int p = e & (P - 1);
  const float* row = teacher + (((long)b * V + fr) * P + p) * D;
  float a = 0.f;
  for (int d = threadIdx.x; d < D; d += 256) { float v = row[d]; a += v * v; }
  float s = block_reduce_sum(a, sm);
  float inv = 1.0f / fmaxf(sqrtf(s), 1e-12f);
  long obase = ((long)b * E + e) * D;
  for (int d = threadIdx.x; d < D; d += 256)
    ext_n_bf[obase + d] = (__bf16)(row[d] * inv);
}

// ---------------- batched bf16 WMMA GEMM: C[b,m,n] = sum_d A[b,m,d]*B[b,n,d] ----------------
// Register-blocked: each wave computes a 32x64 output tile (2 A-frags x 4 B-frags,
// 8 independent f32 accumulators) -> 8 WMMAs per 12 b128 loads per K-step.
__global__ void gemm_bf16_nt(const __bf16* __restrict__ A, const __bf16* __restrict__ Bm,
                             float* __restrict__ C, int N,
                             long sA, long sB, long sC,
                             int tilesM, int tilesN, int totalWaves) {
  int wave = (int)(((long)blockIdx.x * blockDim.x + threadIdx.x) >> 5);
  if (wave >= totalWaves) return;       // wave-uniform: EXEC stays all-ones for WMMA
  int lane = (int)(threadIdx.x & 31);
  int tpb = tilesM * tilesN;
  int b = wave / tpb;
  int t = wave - b * tpb;
  int mt = t / tilesN;
  int nt = t - mt * tilesN;
  int m0 = mt << 5;                     // 32-row block
  int n0 = nt << 6;                     // 64-col block
  int l16 = lane & 15;
  int hi = lane >> 4;
  // A fragment (16x32 bf16): lane<16 holds row, K kb..kb+15;
  // lane>=16 holds the same row's K kb+16..kb+31 (matches B's contiguous K order).
  const __bf16* abase = A + (long)b * sA + (long)(m0 + l16) * D + (hi ? 16 : 0);
  // B fragment (32x16 bf16): lane<16 = column, K kb..kb+15; lane>=16 = K kb+16..kb+31.
  const __bf16* bbase = Bm + (long)b * sB + (long)(n0 + l16) * D + (hi ? 16 : 0);

  v8f acc[2][4];
#pragma unroll
  for (int i = 0; i < 2; i++)
#pragma unroll
    for (int j = 0; j < 4; j++)
      acc[i][j] = (v8f){0.f, 0.f, 0.f, 0.f, 0.f, 0.f, 0.f, 0.f};

#pragma unroll 2
  for (int kb = 0; kb < D; kb += 32) {
    v16bf av[2];
#pragma unroll
    for (int i = 0; i < 2; i++) {
      const __bf16* ap = abase + (long)i * 16 * D + kb;
      v8bf a0 = *(const v8bf*)(ap);
      v8bf a1 = *(const v8bf*)(ap + 8);
#pragma unroll
      for (int q = 0; q < 8; q++) { av[i][q] = a0[q]; av[i][8 + q] = a1[q]; }
    }
    v16bf bv[4];
#pragma unroll
    for (int j = 0; j < 4; j++)
      bv[j] = *(const v16bf*)(bbase + (long)j * 16 * D + kb);
#pragma unroll
    for (int i = 0; i < 2; i++)
#pragma unroll
      for (int j = 0; j < 4; j++)
        acc[i][j] = __builtin_amdgcn_wmma_f32_16x16x32_bf16(
            false, av[i], false, bv[j], (short)0, acc[i][j], false, false);
  }

  // C/D layout: VGPR v -> (M=tile_m+v, N=tile_n+lane) for lanes 0-15;
  //             (M=tile_m+8+v, N=tile_n+lane-16) for lanes 16-31.
#pragma unroll
  for (int i = 0; i < 2; i++) {
#pragma unroll
    for (int j = 0; j < 4; j++) {
      float* crow = C + (long)b * sC
                  + (long)(m0 + i * 16 + (hi << 3)) * N
                  + (n0 + j * 16 + l16);
#pragma unroll
      for (int v = 0; v < 8; v++) crow[(long)v * N] = acc[i][j][v];
    }
  }
}

// ---------------- top-4 per (b,r) row of sim ----------------
__global__ void topk_kernel(const float* __restrict__ sim, int* __restrict__ idx_out) {
  __shared__ float vbuf[256 * TK];
  __shared__ int ibuf[256 * TK];
  int br = blockIdx.x;
  int b = br / R, r = br - b * R;
  const float* row = sim + ((long)b * R + r) * E;
  float v[TK]; int ix[TK];
#pragma unroll
  for (int k = 0; k < TK; k++) { v[k] = -3.4e38f; ix[k] = 0; }
  for (int e = threadIdx.x; e < E; e += 256) {
    float val = row[e];
    if (val > v[TK - 1]) {
      int k = TK - 1;
      while (k > 0 && val > v[k - 1]) { v[k] = v[k - 1]; ix[k] = ix[k - 1]; k--; }
      v[k] = val; ix[k] = e;
    }
  }
#pragma unroll
  for (int k = 0; k < TK; k++) {
    vbuf[threadIdx.x * TK + k] = v[k];
    ibuf[threadIdx.x * TK + k] = ix[k];
  }
  __syncthreads();
  if (threadIdx.x == 0) {
    float bv[TK]; int bi[TK];
#pragma unroll
    for (int k = 0; k < TK; k++) { bv[k] = -3.4e38f; bi[k] = 0; }
    for (int i = 0; i < 256 * TK; i++) {
      float val = vbuf[i];
      if (val > bv[TK - 1]) {
        int k = TK - 1;
        while (k > 0 && val > bv[k - 1]) { bv[k] = bv[k - 1]; bi[k] = bi[k - 1]; k--; }
        bv[k] = val; bi[k] = ibuf[i];
      }
    }
    for (int k = 0; k < TK; k++) idx_out[(b * R + r) * TK + k] = bi[k];
  }
}

// ---------------- gather h = extra_t[idx] (f32-exact norms/dots, bf16 copy for GEMM) ----------------
__global__ void build_h_kernel(const float* __restrict__ teacher,
                               const float* __restrict__ student,
                               const int* __restrict__ ref_perm,
                               const int* __restrict__ idx_in,
                               __bf16* __restrict__ h_bf,
                               float* __restrict__ hh,
                               float* __restrict__ rh_t,
                               float* __restrict__ rh_s) {
  __shared__ float sm[256];
  int id = blockIdx.x;                 // (b*R + r)*TK + k
  int b = id / (R * TK);
  int rem = id - b * R * TK;
  int r = rem / TK;
  int e = idx_in[id];
  int fr = 1 + 2 * (e >> 11);
  int p = e & (P - 1);
  const float* hrow = teacher + (((long)b * V + fr) * P + p) * D;
  int rp = ref_perm[r];
  const float* trow = teacher + (((long)b * V) * P + rp) * D;
  const float* srow = student + (((long)b * 4) * P + rp) * D;
  long obase = (long)id * D;
  float ahh = 0.f, art = 0.f, ars = 0.f;
  for (int d = threadIdx.x; d < D; d += 256) {
    float hv = hrow[d];
    ahh += hv * hv;
    art += hv * trow[d];
    ars += hv * srow[d];
    h_bf[obase + d] = (__bf16)hv;
  }
  float shh = block_reduce_sum(ahh, sm);
  float srt = block_reduce_sum(art, sm);
  float srs = block_reduce_sum(ars, sm);
  if (threadIdx.x == 0) { hh[id] = shh; rh_t[id] = srt; rh_s[id] = srs; }
}

// ---------------- angle loss ----------------
__device__ __forceinline__ void angles3(float rr, float ssv, float hhv,
                                        float srv, float rhv, float shv,
                                        float& a1, float& a2, float& a3) {
  float n_sr = fmaxf(sqrtf(fmaxf(ssv - 2.f * srv + rr, 0.f)), EPSF);
  float n_hr = fmaxf(sqrtf(fmaxf(hhv - 2.f * rhv + rr, 0.f)), EPSF);
  float n_sh = fmaxf(sqrtf(fmaxf(ssv - 2.f * shv + hhv, 0.f)), EPSF);
  a1 = (shv - srv - rhv + rr) / (n_sr * n_hr);
  a2 = (srv - rhv - shv + hhv) / (n_hr * n_sh);
  a3 = (rhv - srv - shv + ssv) / (n_sr * n_sh);
}

__global__ void angle_kernel(const float* __restrict__ sr_t, const float* __restrict__ sr_s,
                             const float* __restrict__ shg_t, const float* __restrict__ shg_s,
                             const float* __restrict__ rr_t, const float* __restrict__ rr_s,
                             const float* __restrict__ ss_t, const float* __restrict__ ss_s,
                             const float* __restrict__ hh, const float* __restrict__ rh_t,
                             const float* __restrict__ rh_s, float* __restrict__ partials) {
  __shared__ float sm[256];
  long gid = (long)blockIdx.x * 256 + threadIdx.x;   // enumerates b*R*S
  int b = (int)(gid / ((long)R * S));
  long rem = gid - (long)b * R * S;
  int r = (int)(rem / S);
  int s = (int)(rem - (long)r * S);
  float rrt = rr_t[b * R + r], rrs = rr_s[b * R + r];
  float sst = ss_t[b * S + s], sss = ss_s[b * S + s];
  long sri = ((long)b * R + r) * S + s;
  float srt = sr_t[sri], srs = sr_s[sri];
  float local = 0.f;
#pragma unroll
  for (int k = 0; k < TK; k++) {
    int id = (b * R + r) * TK + k;
    float hhv = hh[id], rht = rh_t[id], rhs = rh_s[id];
    long shi = ((long)b * (R * TK) + r * TK + k) * S + s;
    float sht = shg_t[shi], shs = shg_s[shi];
    float t1, t2, t3, s1, s2, s3;
    angles3(rrt, sst, hhv, srt, rht, sht, t1, t2, t3);
    angles3(rrs, sss, hhv, srs, rhs, shs, s1, s2, s3);
    local += fabsf(s1 - t1) + fabsf(s2 - t2) + fabsf(s3 - t3);
  }
  float bs = block_reduce_sum(local, sm);
  if (threadIdx.x == 0) partials[blockIdx.x] = bs;
}

__global__ void finalize_kernel(const float* __restrict__ partials, float* __restrict__ out) {
  __shared__ float sm[256];
  float a = 0.f;
  for (int i = threadIdx.x; i < NPART; i += 256) a += partials[i];
  float s = block_reduce_sum(a, sm);
  if (threadIdx.x == 0)
    out[0] = s / (float)((long)BATCH * R * S * TK * NF);   // / 6291456
}

// ---------------- host launch ----------------
extern "C" void kernel_launch(void* const* d_in, const int* in_sizes, int n_in,
                              void* d_out, int out_size, void* d_ws, size_t ws_size,
                              hipStream_t stream) {
  const float* teacher = (const float*)d_in[0];
  const float* student = (const float*)d_in[1];
  const int* ref_perm = (const int*)d_in[2];
  const int* shared_perm = (const int*)d_in[3];
  float* out = (float*)d_out;

  char* ws = (char*)d_ws;
  size_t off = 0;
  auto alloc = [&](size_t bytes) -> void* {
    off = (off + 255) & ~(size_t)255;
    void* pp = (void*)(ws + off);
    off += bytes;
    return pp;
  };

  __bf16* ref_t_bf = (__bf16*)alloc((size_t)BATCH * R * D * 2);
  __bf16* ref_s_bf = (__bf16*)alloc((size_t)BATCH * R * D * 2);
  __bf16* ref_n_bf = (__bf16*)alloc((size_t)BATCH * R * D * 2);
  __bf16* sh_t_bf  = (__bf16*)alloc((size_t)NF * BATCH * S * D * 2);
  __bf16* sh_s_bf  = (__bf16*)alloc((size_t)NF * BATCH * S * D * 2);
  __bf16* ext_n_bf = (__bf16*)alloc((size_t)BATCH * E * D * 2);
  __bf16* h_bf     = (__bf16*)alloc((size_t)BATCH * R * TK * D * 2);
  float* rr_t = (float*)alloc((size_t)BATCH * R * 4);
  float* rr_s = (float*)alloc((size_t)BATCH * R * 4);
  float* ss_t = (float*)alloc((size_t)NF * BATCH * S * 4);
  float* ss_s = (float*)alloc((size_t)NF * BATCH * S * 4);
  float* hh   = (float*)alloc((size_t)BATCH * R * TK * 4);
  float* rh_t = (float*)alloc((size_t)BATCH * R * TK * 4);
  float* rh_s = (float*)alloc((size_t)BATCH * R * TK * 4);
  int*   idxb = (int*)alloc((size_t)BATCH * R * TK * 4);
  float* partials = (float*)alloc((size_t)NPART * 4);
  float* sim = (float*)alloc((size_t)BATCH * R * E * 4);      // 33.5 MB

  // sim is dead after topk; overlay the per-frame GEMM outputs on it (20.97 MB <= 33.5 MB)
  char* simc = (char*)sim;
  float* sr_t  = (float*)simc;
  float* sr_s  = (float*)(simc + (size_t)BATCH * R * S * 4);
  float* shg_t = (float*)(simc + 2 * (size_t)BATCH * R * S * 4);
  float* shg_s = (float*)(simc + 2 * (size_t)BATCH * R * S * 4 + (size_t)BATCH * R * TK * S * 4);

  auto launch_gemm = [&](const __bf16* A, const __bf16* Bm, float* C, int M, int N) {
    int tilesM = M / 32, tilesN = N / 64;   // 32x64 tile per wave
    int totalWaves = BATCH * tilesM * tilesN;
    int blocks = (totalWaves + 7) / 8;
    gemm_bf16_nt<<<blocks, 256, 0, stream>>>(A, Bm, C, N,
        (long)M * D, (long)N * D, (long)M * N, tilesM, tilesN, totalWaves);
  };

  // Stage 1: gathers / conversions / norms
  prep_ref_kernel<<<BATCH * R, 256, 0, stream>>>(teacher, student, ref_perm,
      ref_t_bf, ref_s_bf, ref_n_bf, rr_t, rr_s);
  prep_shared_kernel<<<NF * BATCH * S, 256, 0, stream>>>(teacher, student, shared_perm,
      sh_t_bf, sh_s_bf, ss_t, ss_s);
  prep_ext_kernel<<<BATCH * E, 256, 0, stream>>>(teacher, ext_n_bf);

  // Stage 2: sim GEMM (512 x 8192 x 1024 per batch, bf16 WMMA, f32 accumulate)
  launch_gemm(ref_n_bf, ext_n_bf, sim, R, E);

  // Stage 3: top-4 + gather h rows (exact f32 hh/rh)
  topk_kernel<<<BATCH * R, 256, 0, stream>>>(sim, idxb);
  build_h_kernel<<<BATCH * R * TK, 256, 0, stream>>>(teacher, student, ref_perm, idxb,
      h_bf, hh, rh_t, rh_s);

  // Stage 4: per-frame GEMMs + angle accumulation
  for (int f = 0; f < NF; f++) {
    const __bf16* shtf = sh_t_bf + (size_t)f * BATCH * S * D;
    const __bf16* shsf = sh_s_bf + (size_t)f * BATCH * S * D;
    launch_gemm(ref_t_bf, shtf, sr_t, R, S);
    launch_gemm(ref_s_bf, shsf, sr_s, R, S);
    launch_gemm(h_bf, shtf, shg_t, R * TK, S);
    launch_gemm(h_bf, shsf, shg_s, R * TK, S);
    angle_kernel<<<ANGLE_BLOCKS, 256, 0, stream>>>(sr_t, sr_s, shg_t, shg_s,
        rr_t, rr_s, ss_t + f * BATCH * S, ss_s + f * BATCH * S,
        hh, rh_t, rh_s, partials + f * ANGLE_BLOCKS);
  }

  // Stage 5: deterministic final reduction
  finalize_kernel<<<1, 256, 0, stream>>>(partials, out);
}
